// SelfAttention1d_36146444763993
// MI455X (gfx1250) — matrix operand order repaired and checked by reading, not compiled, not run on hardware
//
#include <hip/hip_runtime.h>
#include <hip/hip_bf16.h>
#include <stdint.h>

// ---------------------------------------------------------------------------
// SelfAttention1d for MI455X (gfx1250): bf16 WMMA everywhere, flash-style
// softmax fusion (never materializes the [B,1024,2048] score matrix).
// B=16, C=512, W=2048.  GEMM stages use 2x2 register blocking (32x32 per
// wave) to double arithmetic intensity vs. L2.
// ---------------------------------------------------------------------------

typedef __bf16 bf16_t;
typedef __attribute__((ext_vector_type(16))) __bf16 v16bf;
typedef __attribute__((ext_vector_type(8)))  float  v8f;

union FragU { v16bf v; uint4 q[2]; };
union Pack8 { bf16_t e[8]; uint4 q; };

__device__ __forceinline__ v8f wmma_bf16(v16bf a, v16bf b, v8f c) {
  // D = A(16x32 bf16) * B(32x16 bf16) + C(16x16 f32)
  return __builtin_amdgcn_wmma_f32_16x16x32_bf16(
      /*neg_a=*/false, a, /*neg_b=*/false, b,
      /*c_mod=*/(short)0, c, /*reuse_a=*/false, /*reuse_b=*/false);
}

// A-fragment: 16x32 (MxK) from row-major [M x rs] bf16.  base -> (m0,k0).
// Lanes 0-15 hold rows 0-15; VGPR0-3: K = 8h+0..7, VGPR4-7: K = 16+8h+0..7.
__device__ __forceinline__ v16bf load_fragA(const bf16_t* base, int rs, int lane) {
  int m = lane & 15, h = lane >> 4;
  const bf16_t* r = base + (size_t)m * rs + h * 8;
  FragU f;
  f.q[0] = *(const uint4*)(r);        // K = 8h .. 8h+7
  f.q[1] = *(const uint4*)(r + 16);   // K = 16+8h .. 16+8h+7
  return f.v;
}

// B-fragment: 32x16 (KxN) from TRANSPOSED storage [N x rs] bf16 (row n = col n
// of B, contiguous in K).  baseT -> (n0,k0).  Lane n holds K = 8h..8h+7, 16+8h..16+8h+7.
__device__ __forceinline__ v16bf load_fragB_T(const bf16_t* baseT, int rs, int lane) {
  int n = lane & 15, h = lane >> 4;
  const bf16_t* r = baseT + (size_t)n * rs + h * 8;
  FragU f;
  f.q[0] = *(const uint4*)(r);
  f.q[1] = *(const uint4*)(r + 16);
  return f.v;
}

// ---------------- stage 0: transpose x to bf16 [B, 2048, 512] ---------------
__global__ void k_transpose_x(const float* __restrict__ x, bf16_t* __restrict__ xT) {
  __shared__ float t[32][33];
  int b = blockIdx.z;
  int c0 = blockIdx.y * 32, w0 = blockIdx.x * 32;
  int tx = threadIdx.x, ty = threadIdx.y;               // block (32,8)
  const float* xb = x + (size_t)b * 512 * 2048;
#pragma unroll
  for (int i = 0; i < 32; i += 8)
    t[ty + i][tx] = xb[(size_t)(c0 + ty + i) * 2048 + w0 + tx];
  __syncthreads();
  bf16_t* xtb = xT + ((size_t)b * 2048 + w0) * 512;
#pragma unroll
  for (int i = 0; i < 32; i += 8)
    xtb[(size_t)(ty + i) * 512 + c0 + tx] = (bf16_t)t[tx][ty + i];
}

__global__ void k_f2bf(const float* __restrict__ s, bf16_t* __restrict__ d, int n) {
  int i = blockIdx.x * blockDim.x + threadIdx.x;
  if (i < n) d[i] = (bf16_t)s[i];
}

// ------------- stage 1: conv1x1 as GEMM, transposed bf16 output -------------
// out[Cout,W] = Wm[Cout,512] @ x[512,W] + bias  ->  outT[B, 2048, M] (bf16)
// 2x2 register blocking: one wave computes a 32x32 output tile.
__device__ __forceinline__ void store_tileT(bf16_t* __restrict__ outT, int M,
                                            size_t row, int mcol, int n, int h,
                                            const float* __restrict__ bias, v8f acc) {
  Pack8 u;
  const float* bp = bias + mcol + 8 * h;
#pragma unroll
  for (int r = 0; r < 8; ++r) u.e[r] = (bf16_t)(acc[r] + bp[r]);
  *(uint4*)(outT + (row + n) * M + mcol + 8 * h) = u.q;
}

__global__ void k_conv_gemm(const bf16_t* __restrict__ Wm,   // [M,512]
                            const float*  __restrict__ bias, // [M]
                            const bf16_t* __restrict__ xT,   // [B,2048,512]
                            bf16_t*       __restrict__ outT, // [B,2048,M]
                            int M) {
  int wid  = (blockIdx.x * blockDim.x + threadIdx.x) >> 5;
  int lane = threadIdx.x & 31;
  const int nt2 = 64;                     // 2048/32
  int mt2 = M >> 5;
  int b   = wid / (nt2 * mt2);
  int rem = wid % (nt2 * mt2);
  int nt  = rem / mt2, mt = rem % mt2;
  int m0 = mt * 32, n0 = nt * 32;
  const bf16_t* xb = xT + ((size_t)b * 2048 + n0) * 512;
  const bf16_t* w0p = Wm + (size_t)m0 * 512;
  v8f acc00 = {}, acc01 = {}, acc10 = {}, acc11 = {};
#pragma unroll
  for (int k0 = 0; k0 < 512; k0 += 32) {
    v16bf a0 = load_fragA(w0p + k0, 512, lane);
    v16bf a1 = load_fragA(w0p + (size_t)16 * 512 + k0, 512, lane);
    v16bf b0 = load_fragB_T(xb + k0, 512, lane);
    v16bf b1 = load_fragB_T(xb + (size_t)16 * 512 + k0, 512, lane);
    acc00 = wmma_bf16(a0, b0, acc00);
    acc01 = wmma_bf16(a0, b1, acc01);
    acc10 = wmma_bf16(a1, b0, acc10);
    acc11 = wmma_bf16(a1, b1, acc11);
  }
  int n = lane & 15, h = lane >> 4;
  size_t row = (size_t)b * 2048 + n0;
  store_tileT(outT, M, row,      m0,      n, h, bias, acc00);
  store_tileT(outT, M, row + 16, m0,      n, h, bias, acc01);
  store_tileT(outT, M, row,      m0 + 16, n, h, bias, acc10);
  store_tileT(outT, M, row + 16, m0 + 16, n, h, bias, acc11);
}

// ------------------- stage 2: maxpool2 of conv_f (bf16) ---------------------
__global__ void k_pool_f(const bf16_t* __restrict__ cfT,  // [B,2048,128]
                         bf16_t* __restrict__ fT) {       // [B,1024,128]
  size_t i = (size_t)blockIdx.x * blockDim.x + threadIdx.x;
  if (i >= ((size_t)16 << 17)) return;
  int c = i & 127; int q = (i >> 7) & 1023; int b = (int)(i >> 17);
  const bf16_t* src = cfT + ((size_t)b * 2048 + 2 * q) * 128 + c;
  fT[i] = (bf16_t)fmaxf((float)src[0], (float)src[128]);
}

// --------- stage 3: per-row softmax stats (m_q, l_q) via WMMA + shfl --------
__global__ void k_row_stats(const bf16_t* __restrict__ fT,  // [B,1024,128]
                            const bf16_t* __restrict__ gT,  // [B,2048,128]
                            float* __restrict__ rowM,       // [B,1024]
                            float* __restrict__ rowL) {
  int b = blockIdx.y, q0 = blockIdx.x * 16;
  int wave = threadIdx.x >> 5, lane = threadIdx.x & 31;     // 8 waves
  __shared__ float pm[8][16], pl[8][16];
  const bf16_t* fb = fT + ((size_t)b * 1024 + q0) * 128;
  const bf16_t* gb = gT + (size_t)b * 2048 * 128;
  float mx[8], l[8];
#pragma unroll
  for (int r = 0; r < 8; ++r) { mx[r] = -3.0e38f; l[r] = 0.f; }
  for (int kt = 0; kt < 16; ++kt) {                         // 256 k per wave
    int k0 = wave * 256 + kt * 16;
    v8f s = {};
#pragma unroll
    for (int c0 = 0; c0 < 128; c0 += 32) {
      v16bf a  = load_fragA(fb + c0, 128, lane);
      v16bf bm = load_fragB_T(gb + (size_t)k0 * 128 + c0, 128, lane);
      s = wmma_bf16(a, bm, s);
    }
#pragma unroll
    for (int r = 0; r < 8; ++r) {                           // online softmax
      float v = s[r];
      if (v > mx[r]) { l[r] = l[r] * __expf(mx[r] - v) + 1.f; mx[r] = v; }
      else           { l[r] += __expf(v - mx[r]); }
    }
  }
  // combine across the 16 lanes that share the same 8 rows (xor stays in half)
#pragma unroll
  for (int d = 1; d < 16; d <<= 1) {
#pragma unroll
    for (int r = 0; r < 8; ++r) {
      float om = __shfl_xor(mx[r], d, 32);
      float ol = __shfl_xor(l[r],  d, 32);
      float nm = fmaxf(mx[r], om);
      l[r] = l[r] * __expf(mx[r] - nm) + ol * __expf(om - nm);
      mx[r] = nm;
    }
  }
  if ((lane & 15) == 0) {
    int h = lane >> 4;
#pragma unroll
    for (int r = 0; r < 8; ++r) { pm[wave][8 * h + r] = mx[r]; pl[wave][8 * h + r] = l[r]; }
  }
  __syncthreads();
  if (threadIdx.x < 16) {
    float m = pm[0][threadIdx.x], ll = pl[0][threadIdx.x];
#pragma unroll
    for (int wv = 1; wv < 8; ++wv) {
      float om = pm[wv][threadIdx.x], ol = pl[wv][threadIdx.x];
      float nm = fmaxf(m, om);
      ll = ll * __expf(m - nm) + ol * __expf(om - nm);
      m = nm;
    }
    rowM[(size_t)b * 1024 + q0 + threadIdx.x] = m;
    rowL[(size_t)b * 1024 + q0 + threadIdx.x] = ll;
  }
}

// ------ stage 4: h' = maxpool2(conv_h) / l_q, laid out [B, 256, 1024] -------
__global__ void k_hprep(const bf16_t* __restrict__ chT,  // [B,2048,256]
                        const float* __restrict__ rowL,
                        bf16_t* __restrict__ hp) {       // [B,256,1024]
  size_t i = (size_t)blockIdx.x * blockDim.x + threadIdx.x;
  if (i >= ((size_t)16 << 18)) return;
  int q = i & 1023; int c = (i >> 10) & 255; int b = (int)(i >> 18);
  const bf16_t* src = chT + ((size_t)b * 2048 + 2 * q) * 256 + c;
  float v = fmaxf((float)src[0], (float)src[256]);
  hp[i] = (bf16_t)(v / rowL[(size_t)b * 1024 + q]);
}

// ------ stage 5: o = h' * exp(s - m_q), s recomputed in-register ------------
// Block = 8 waves, one 16-wide k tile, full C/2=256.  Waves 0-1 produce the
// 32x16 p-tile into LDS (B-fragment friendly [k][q] layout); all 8 waves
// consume it for 2 c-tiles each.  Output stored transposed bf16 [B,2048,256].
__global__ void k_attn_out(const bf16_t* __restrict__ fT,  // [B,1024,128]
                           const bf16_t* __restrict__ gT,  // [B,2048,128]
                           const bf16_t* __restrict__ hp,  // [B,256,1024]
                           const float*  __restrict__ rowM,
                           bf16_t*       __restrict__ oT) { // [B,2048,256]
  int b = blockIdx.y, k0 = blockIdx.x * 16;
  int wave = threadIdx.x >> 5, lane = threadIdx.x & 31;
  int n = lane & 15, h = lane >> 4;
  __shared__ bf16_t pT[16][32];                       // [k-local][q-local]
  const bf16_t* fb = fT + (size_t)b * 1024 * 128;
  const bf16_t* gb = gT + ((size_t)b * 2048 + k0) * 128;
  const bf16_t* hb = hp + (size_t)b * 256 * 1024;
  const float*  mrow = rowM + (size_t)b * 1024;
  v8f acc0 = {}, acc1 = {};
  for (int q0 = 0; q0 < 1024; q0 += 32) {
    if (wave < 2) {                                   // producers
      v8f s = {};
      const bf16_t* fq = fb + (size_t)(q0 + wave * 16) * 128;
#pragma unroll
      for (int c0 = 0; c0 < 128; c0 += 32) {
        v16bf a  = load_fragA(fq + c0, 128, lane);
        v16bf bm = load_fragB_T(gb + c0, 128, lane);
        s = wmma_bf16(a, bm, s);
      }
      const float* mq = mrow + q0 + wave * 16 + 8 * h; // rows 8h..8h+7
      Pack8 u;
#pragma unroll
      for (int r = 0; r < 8; ++r) u.e[r] = (bf16_t)__expf(s[r] - mq[r]);
      *(uint4*)&pT[n][wave * 16 + 8 * h] = u.q;
    }
    __syncthreads();
    {
      int m0 = wave * 32;
      v16bf a0 = load_fragA(hb + (size_t)m0 * 1024 + q0, 1024, lane);
      v16bf a1 = load_fragA(hb + (size_t)(m0 + 16) * 1024 + q0, 1024, lane);
      v16bf bm = load_fragB_T(&pT[0][0], 32, lane);
      acc0 = wmma_bf16(a0, bm, acc0);
      acc1 = wmma_bf16(a1, bm, acc1);
    }
    __syncthreads();
  }
  int m0 = wave * 32;
  Pack8 u0, u1;
#pragma unroll
  for (int r = 0; r < 8; ++r) { u0.e[r] = (bf16_t)acc0[r]; u1.e[r] = (bf16_t)acc1[r]; }
  bf16_t* orow = oT + ((size_t)b * 2048 + k0 + n) * 256;
  *(uint4*)(orow + m0 + 8 * h)      = u0.q;
  *(uint4*)(orow + m0 + 16 + 8 * h) = u1.q;
}

// ------- stage 6: out = gamma * (wa @ o + ba) + x  (fp32 output) ------------
// 2x2 register blocking: one wave computes a 32x32 output tile.
__global__ void k_final(const bf16_t* __restrict__ wa,  // [512,256]
                        const float*  __restrict__ ba,  // [512]
                        const bf16_t* __restrict__ oT,  // [B,2048,256]
                        const float*  __restrict__ x,   // [B,512,2048]
                        const float*  __restrict__ gamma,
                        float* __restrict__ out) {      // [B,512,2048]
  int wid  = (blockIdx.x * blockDim.x + threadIdx.x) >> 5;
  int lane = threadIdx.x & 31;
  const int mt2 = 16, nt2 = 64;           // 512/32, 2048/32
  int b   = wid / (mt2 * nt2);
  int rem = wid % (mt2 * nt2);
  int nt  = rem / mt2, mt = rem % mt2;
  int m0 = mt * 32, n0 = nt * 32;
  const bf16_t* ob = oT + ((size_t)b * 2048 + n0) * 256;
  const bf16_t* wp = wa + (size_t)m0 * 256;
  v8f acc00 = {}, acc01 = {}, acc10 = {}, acc11 = {};
#pragma unroll
  for (int k0 = 0; k0 < 256; k0 += 32) {
    v16bf a0 = load_fragA(wp + k0, 256, lane);
    v16bf a1 = load_fragA(wp + (size_t)16 * 256 + k0, 256, lane);
    v16bf b0 = load_fragB_T(ob + k0, 256, lane);
    v16bf b1 = load_fragB_T(ob + (size_t)16 * 256 + k0, 256, lane);
    acc00 = wmma_bf16(a0, b0, acc00);
    acc01 = wmma_bf16(a0, b1, acc01);
    acc10 = wmma_bf16(a1, b0, acc10);
    acc11 = wmma_bf16(a1, b1, acc11);
  }
  int n = lane & 15, h = lane >> 4;
  float gm = gamma[0];
  const float* xb  = x   + (size_t)b * 512 * 2048;
  float*       odb = out + (size_t)b * 512 * 2048;
#pragma unroll
  for (int r = 0; r < 8; ++r) {
    int ma = m0 + 8 * h + r, mb = ma + 16;
    size_t i00 = (size_t)ma * 2048 + n0 + n;
    size_t i01 = i00 + 16;
    size_t i10 = (size_t)mb * 2048 + n0 + n;
    size_t i11 = i10 + 16;
    odb[i00] = gm * (acc00[r] + ba[ma]) + xb[i00];
    odb[i01] = gm * (acc01[r] + ba[ma]) + xb[i01];
    odb[i10] = gm * (acc10[r] + ba[mb]) + xb[i10];
    odb[i11] = gm * (acc11[r] + ba[mb]) + xb[i11];
  }
}

// ---------------------------------------------------------------------------
extern "C" void kernel_launch(void* const* d_in, const int* in_sizes, int n_in,
                              void* d_out, int out_size, void* d_ws, size_t ws_size,
                              hipStream_t stream) {
  const float* x     = (const float*)d_in[0];
  const float* wf    = (const float*)d_in[1];
  const float* bf    = (const float*)d_in[2];
  const float* wg    = (const float*)d_in[3];
  const float* bg    = (const float*)d_in[4];
  const float* wh    = (const float*)d_in[5];
  const float* bh    = (const float*)d_in[6];
  const float* wa    = (const float*)d_in[7];
  const float* ba    = (const float*)d_in[8];
  const float* gamma = (const float*)d_in[9];
  float* out = (float*)d_out;

  // bump allocator over d_ws (~97 MB total)
  char* p = (char*)d_ws;
  auto alloc = [&](size_t bytes) { char* r = p; p += (bytes + 255) & ~size_t(255); return r; };
  bf16_t* xT    = (bf16_t*)alloc((size_t)16 * 2048 * 512 * 2);  // 32 MB
  bf16_t* wfb   = (bf16_t*)alloc((size_t)128 * 512 * 2);
  bf16_t* wgb   = (bf16_t*)alloc((size_t)128 * 512 * 2);
  bf16_t* whb   = (bf16_t*)alloc((size_t)256 * 512 * 2);
  bf16_t* wab   = (bf16_t*)alloc((size_t)512 * 256 * 2);
  bf16_t* cfT   = (bf16_t*)alloc((size_t)16 * 2048 * 128 * 2);  // 8 MB
  bf16_t* gTbuf = (bf16_t*)alloc((size_t)16 * 2048 * 128 * 2);  // 8 MB
  bf16_t* chT   = (bf16_t*)alloc((size_t)16 * 2048 * 256 * 2);  // 16 MB
  bf16_t* fTbuf = (bf16_t*)alloc((size_t)16 * 1024 * 128 * 2);  // 4 MB
  bf16_t* hpbuf = (bf16_t*)alloc((size_t)16 * 256 * 1024 * 2);  // 8 MB
  float*  rowM  = (float*)alloc((size_t)16 * 1024 * 4);
  float*  rowL  = (float*)alloc((size_t)16 * 1024 * 4);
  bf16_t* oTbuf = (bf16_t*)alloc((size_t)16 * 2048 * 256 * 2);  // 16 MB
  (void)ws_size; (void)in_sizes; (void)n_in; (void)out_size;

  // stage 0: transpose/convert
  k_transpose_x<<<dim3(64, 16, 16), dim3(32, 8), 0, stream>>>(x, xT);
  k_f2bf<<<(128 * 512 + 255) / 256, 256, 0, stream>>>(wf, wfb, 128 * 512);
  k_f2bf<<<(128 * 512 + 255) / 256, 256, 0, stream>>>(wg, wgb, 128 * 512);
  k_f2bf<<<(256 * 512 + 255) / 256, 256, 0, stream>>>(wh, whb, 256 * 512);
  k_f2bf<<<(512 * 256 + 255) / 256, 256, 0, stream>>>(wa, wab, 512 * 256);

  // stage 1: the three input convs (one wave per 32x32 tile, 8 waves/block)
  k_conv_gemm<<<16 * 64 * 4 / 8, 256, 0, stream>>>(wfb, bf, xT, cfT, 128);
  k_conv_gemm<<<16 * 64 * 4 / 8, 256, 0, stream>>>(wgb, bg, xT, gTbuf, 128);
  k_conv_gemm<<<16 * 64 * 8 / 8, 256, 0, stream>>>(whb, bh, xT, chT, 256);

  // stage 2: pool f
  k_pool_f<<<(int)(((size_t)16 << 17) / 256), 256, 0, stream>>>(cfT, fTbuf);

  // stage 3: softmax row statistics
  k_row_stats<<<dim3(64, 16), 256, 0, stream>>>(fTbuf, gTbuf, rowM, rowL);

  // stage 4: h' = pool(conv_h)/l_q
  k_hprep<<<(int)(((size_t)16 << 18) / 256), 256, 0, stream>>>(chT, rowL, hpbuf);

  // stage 5: fused attention output (s recompute + exp + WMMA via LDS)
  k_attn_out<<<dim3(128, 16), 256, 0, stream>>>(fTbuf, gTbuf, hpbuf, rowM, oTbuf);

  // stage 6: final conv + residual (one wave per 32x32 tile)
  k_final<<<16 * 64 * 16 / 8, 256, 0, stream>>>(wab, ba, oTbuf, x, gamma, out);
}
